// LineageLinkPredictionGNN_37271726195066
// MI455X (gfx1250) — matrix operand order repaired and checked by reading, not compiled, not run on hardware
//
#include <hip/hip_runtime.h>
#include <hip/hip_bf16.h>

#define N_NODES 50000
#define N_EDGES 800000
#define IN_C    32
#define EDGE_C  16
#define H       64
#define BN_EPS  1e-5f
#define COS_EPS 1e-8f
#define KPAD    288   // 257 real K for ee1, padded to 9 chunks of 32

typedef __attribute__((ext_vector_type(16))) __bf16 v16bf;
typedef __attribute__((ext_vector_type(8)))  float  v8f;

__device__ __forceinline__ float relu_f(float x) { return x > 0.f ? x : 0.f; }

// scalar |x| via integer mask -- avoids the packed-fabs/bf16-cvt isel crash
__device__ __forceinline__ float abs_bits(float x) {
  union { float f; unsigned u; } d;
  d.f = x;
  d.u &= 0x7fffffffu;
  return d.f;
}

// A fragment (16-bit, 16x32 MxK): lanes 0-15 hold M=0..15 / K base 0, lanes 16-31 K base +16;
// elems 0..7 -> K = kh+0..7, elems 8..15 -> K = kh+8..15.
// A is row-major in LDS, so each half-fragment is 8 contiguous bf16 -> ds_load_b128.
__device__ __forceinline__ v16bf load_a_frag(const __bf16* A, int lda, int kbase, int lane) {
  int m  = lane & 15;
  int kh = (lane >> 4) << 4;
  v16bf a;
#pragma unroll
  for (int j = 0; j < 8; ++j) {
    a[j]     = A[m * lda + kbase + kh + j];
    a[j + 8] = A[m * lda + kbase + 8 + kh + j];
  }
  return a;
}

// B fragments are stored in LDS pre-swizzled into fragment order:
// within one 32xN chunk, element (krel, n) lives at
//   [ ((n>>4)*32 + ((n&15) | 16*(krel>>4))) * 16 + (krel&15) ]
// so each lane reads 16 contiguous bf16 (two ds_load_b128).
__device__ __forceinline__ int bswz(int krel, int n) {
  return (((n >> 4) * 32) + ((n & 15) | (((krel >> 4) & 1) << 4))) * 16 + (krel & 15);
}
__device__ __forceinline__ v16bf load_b_frag_sw(const __bf16* Wchunk, int coltile, int lane) {
  const __bf16* p = Wchunk + (coltile * 32 + lane) * 16;
  v16bf b;
#pragma unroll
  for (int j = 0; j < 16; ++j) b[j] = p[j];
  return b;
}

__device__ __forceinline__ v8f wmma_bf16(v16bf a, v16bf b, v8f c) {
  return __builtin_amdgcn_wmma_f32_16x16x32_bf16(false, a, false, b, (short)0, c, false, false);
}

// ---------------- zero ----------------
__global__ void zero_kernel(float* __restrict__ p, long long n) {
  for (long long i = (long long)blockIdx.x * blockDim.x + threadIdx.x; i < n;
       i += (long long)gridDim.x * blockDim.x)
    p[i] = 0.f;
}

// ---------------- node_proj: x[N,32] @ W[32,64] + b, relu ----------------
__global__ __launch_bounds__(128)
void node_proj_kernel(const float* __restrict__ X, const float* __restrict__ W,
                      const float* __restrict__ B, float* __restrict__ out) {
  __shared__ __bf16 Alds[16 * 32];
  __shared__ __bf16 Wlds[32 * 64];   // 1 chunk, 4 column tiles, swizzled
  int t = threadIdx.x, lane = t & 31, wave = t >> 5;
  int row0 = blockIdx.x * 16;
  for (int i = t; i < 16 * 32; i += 128) {
    int r = i >> 5, c = i & 31;
    Alds[i] = (__bf16)X[(row0 + r) * IN_C + c];
  }
  for (int i = t; i < 32 * 64; i += 128) {
    int kr = i >> 6, n = i & 63;
    Wlds[bswz(kr, n)] = (__bf16)W[i];
  }
  __syncthreads();
  v8f acc = {};
  v16bf a = load_a_frag(Alds, 32, 0, lane);
  v16bf b = load_b_frag_sw(Wlds, wave, lane);
  acc = wmma_bf16(a, b, acc);
  int n = lane & 15, kh = lane >> 4;
  int c = wave * 16 + n;
  float bias = B[c];
#pragma unroll
  for (int r = 0; r < 8; ++r) out[(row0 + r + 8 * kh) * H + c] = relu_f(acc[r] + bias);
}

// ---------------- edge_proj: ea[E,16] @ W[16,64] + b, relu ----------------
__global__ __launch_bounds__(128)
void edge_proj_kernel(const float* __restrict__ EA, const float* __restrict__ W,
                      const float* __restrict__ B, float* __restrict__ out) {
  __shared__ __bf16 Alds[16 * 32];
  __shared__ __bf16 Wlds[32 * 64];
  int t = threadIdx.x, lane = t & 31, wave = t >> 5;
  int e0 = blockIdx.x * 16;
  for (int i = t; i < 16 * 32; i += 128) {
    int r = i >> 5, c = i & 31;
    Alds[i] = (c < EDGE_C) ? (__bf16)EA[(e0 + r) * EDGE_C + c] : (__bf16)0.f;
  }
  for (int i = t; i < 32 * 64; i += 128) {
    int kr = i >> 6, n = i & 63;
    Wlds[bswz(kr, n)] = (kr < EDGE_C) ? (__bf16)W[kr * 64 + n] : (__bf16)0.f;
  }
  __syncthreads();
  v8f acc = {};
  v16bf a = load_a_frag(Alds, 32, 0, lane);
  v16bf b = load_b_frag_sw(Wlds, wave, lane);
  acc = wmma_bf16(a, b, acc);
  int n = lane & 15, kh = lane >> 4;
  int c = wave * 16 + n;
  float bias = B[c];
#pragma unroll
  for (int r = 0; r < 8; ++r) out[(e0 + r + 8 * kh) * H + c] = relu_f(acc[r] + bias);
}

// ---------------- pdn_e1 + pdn_e2 fused: omega[e] ----------------
__global__ __launch_bounds__(64)
void pdn_edge_kernel(const float* __restrict__ EH, const float* __restrict__ W1,
                     const float* __restrict__ B1, const float* __restrict__ W2,
                     const float* __restrict__ B2, float* __restrict__ omega) {
  __shared__ __bf16 Alds[16 * 64];
  __shared__ __bf16 Wlds[64 * 32];   // 2 chunks x 2 column tiles, swizzled (chunk = 1024)
  __shared__ float  Hlds[16 * 32];
  __shared__ float  w2s[32];
  int t = threadIdx.x, lane = t & 31, wave = t >> 5;
  int e0 = blockIdx.x * 16;
  for (int i = t; i < 16 * 64; i += 64) Alds[i] = (__bf16)EH[e0 * H + i];
  for (int i = t; i < 64 * 32; i += 64) {
    int k = i >> 5, n = i & 31;
    Wlds[(k >> 5) * 1024 + bswz(k & 31, n)] = (__bf16)W1[i];
  }
  if (t < 32) w2s[t] = W2[t];
  __syncthreads();
  v8f acc = {};
#pragma unroll
  for (int kc = 0; kc < 2; ++kc) {
    v16bf a = load_a_frag(Alds, 64, kc * 32, lane);
    v16bf b = load_b_frag_sw(Wlds + kc * 1024, wave, lane);
    acc = wmma_bf16(a, b, acc);
  }
  int n = lane & 15, kh = lane >> 4;
  int c = wave * 16 + n;
  float bias = B1[c];
#pragma unroll
  for (int r = 0; r < 8; ++r) Hlds[(r + 8 * kh) * 32 + c] = relu_f(acc[r] + bias);
  __syncthreads();
  if (t < 16) {
    float s = B2[0];
#pragma unroll 4
    for (int c2 = 0; c2 < 32; ++c2) s += Hlds[t * 32 + c2] * w2s[c2];
    omega[e0 + t] = s;
  }
}

// ---------------- pdn_n on gathered x[row], scaled by omega, scatter to agg[col] ----------------
__global__ __launch_bounds__(128)
void pdn_node_scatter_kernel(const float* __restrict__ Xh, const int* __restrict__ row,
                             const int* __restrict__ col, const float* __restrict__ Wn,
                             const float* __restrict__ Bn, const float* __restrict__ omega,
                             float* __restrict__ agg) {
  __shared__ __bf16 Alds[16 * 64];
  __shared__ __bf16 Wlds[64 * 64];   // 2 chunks x 4 column tiles, swizzled (chunk = 2048)
  __shared__ int rowi[16], coli[16];
  __shared__ float om[16];
  int t = threadIdx.x, lane = t & 31, wave = t >> 5;
  int e0 = blockIdx.x * 16;
  if (t < 16) { rowi[t] = row[e0 + t]; coli[t] = col[e0 + t]; om[t] = omega[e0 + t]; }
  __syncthreads();
  for (int i = t; i < 16 * 64; i += 128) {
    int e = i >> 6, c = i & 63;
    Alds[i] = (__bf16)Xh[rowi[e] * H + c];
  }
  for (int i = t; i < 64 * 64; i += 128) {
    int k = i >> 6, n = i & 63;
    Wlds[(k >> 5) * 2048 + bswz(k & 31, n)] = (__bf16)Wn[i];
  }
  __syncthreads();
  v8f acc = {};
#pragma unroll
  for (int kc = 0; kc < 2; ++kc) {
    v16bf a = load_a_frag(Alds, 64, kc * 32, lane);
    v16bf b = load_b_frag_sw(Wlds + kc * 2048, wave, lane);
    acc = wmma_bf16(a, b, acc);
  }
  int n = lane & 15, kh = lane >> 4;
  int c = wave * 16 + n;
  float bias = Bn[c];
#pragma unroll
  for (int r = 0; r < 8; ++r) {
    int m = r + 8 * kh;
    float v = relu_f(acc[r] + bias) * om[m];
    atomicAdd(&agg[coli[m] * H + c], v);
  }
}

// ---------------- s = x + agg; channel sum/sumsq partials ----------------
__global__ __launch_bounds__(256)
void node_sum_stats_kernel(const float* __restrict__ Xh, const float* __restrict__ agg,
                           float* __restrict__ sbuf, float* __restrict__ stats) {
  __shared__ float ssum[H], ssq[H];
  int t = threadIdx.x;
  if (t < H) { ssum[t] = 0.f; ssq[t] = 0.f; }
  __syncthreads();
  const long long total = (long long)N_NODES * H;
  int c = t & 63;  // stride is a multiple of 64, so channel is fixed per thread
  float lsum = 0.f, lsq = 0.f;
  for (long long i = (long long)blockIdx.x * 256 + t; i < total; i += (long long)gridDim.x * 256) {
    float s = Xh[i] + agg[i];
    sbuf[i] = s;
    lsum += s; lsq += s * s;
  }
  atomicAdd(&ssum[c], lsum);
  atomicAdd(&ssq[c], lsq);
  __syncthreads();
  if (t < H) { atomicAdd(&stats[t], ssum[t]); atomicAdd(&stats[H + t], ssq[t]); }
}

// ---------------- BN finalize: mean, rsqrt(var+eps) ----------------
__global__ void bn_finalize_kernel(const float* __restrict__ sums, float count,
                                   float* __restrict__ meaninv) {
  int t = threadIdx.x;
  if (t < H) {
    float m = sums[t] / count;
    float v = sums[H + t] / count - m * m;
    meaninv[t]     = m;
    meaninv[H + t] = rsqrtf(v + BN_EPS);
  }
}

// ---------------- node BN apply + relu -> x_new ----------------
__global__ __launch_bounds__(256)
void node_bn_apply_kernel(const float* __restrict__ sbuf, const float* __restrict__ mi,
                          const float* __restrict__ g, const float* __restrict__ b,
                          float* __restrict__ xout) {
  const long long total = (long long)N_NODES * H;
  for (long long i = (long long)blockIdx.x * 256 + threadIdx.x; i < total;
       i += (long long)gridDim.x * 256) {
    int c = (int)(i & 63);
    xout[i] = relu_f(g[c] * (sbuf[i] - mi[c]) * mi[H + c] + b[c]);
  }
}

// ---------------- fused edge MLP: assemble [16,257] A-tile, ee1 (K=257 -> 128), ee2 (128 -> 64),
//                  write pre-BN e, accumulate edge BN stats ----------------
__global__ __launch_bounds__(256)
void edge_emb_kernel(const float* __restrict__ EH, const float* __restrict__ Xn,
                     const int* __restrict__ row, const int* __restrict__ col,
                     const float* __restrict__ W1, const float* __restrict__ B1,
                     const float* __restrict__ W2, const float* __restrict__ B2,
                     float* __restrict__ epre, float* __restrict__ stats) {
  __shared__ __bf16 Alds[16 * KPAD];
  __shared__ __bf16 Wlds[32 * 128];   // one swizzled 32-K chunk (8 col tiles)
  __shared__ __bf16 Hlds[16 * 128];
  __shared__ int rowi[16], coli[16];
  __shared__ float esum[H], esq[H];
  int t = threadIdx.x, lane = t & 31, wave = t >> 5;
  int e0 = blockIdx.x * 16;
  if (t < 16) { rowi[t] = row[e0 + t]; coli[t] = col[e0 + t]; }
  if (t < H)  { esum[t] = 0.f; esq[t] = 0.f; }
  // zero K padding columns 256..287 (col 256 = cos, rewritten below)
  for (int i = t; i < 16 * (KPAD - 256); i += 256) {
    int e = i / (KPAD - 256), c = 256 + (i % (KPAD - 256));
    Alds[e * KPAD + c] = (__bf16)0.f;
  }
  // edge_attr -> K 0..63
  for (int i = t; i < 16 * 64; i += 256) {
    int e = i >> 6, c = i & 63;
    Alds[e * KPAD + c] = (__bf16)EH[(e0 + e) * H + c];
  }
  __syncthreads();
  // src / tgt / |src-tgt| + cosine; 16 threads per edge (contiguous lanes).
  // Channels strided by 16 per thread: keeps bf16 LDS stores non-adjacent so the
  // SLP vectorizer cannot fuse packed fabs + packed bf16-convert (backend isel bug).
  {
    int e  = t >> 4;
    int k  = t & 15;
    const float* srcp = Xn + (long long)rowi[e] * H;
    const float* tgtp = Xn + (long long)coli[e] * H;
    float dot = 0.f, ns = 0.f, nt = 0.f;
#pragma unroll
    for (int j = 0; j < 4; ++j) {
      int c = k + j * 16;
      float s = srcp[c], tg = tgtp[c];
      Alds[e * KPAD + 64 + c]  = (__bf16)s;
      Alds[e * KPAD + 128 + c] = (__bf16)tg;
      Alds[e * KPAD + 192 + c] = (__bf16)abs_bits(s - tg);
      dot += s * tg; ns += s * s; nt += tg * tg;
    }
#pragma unroll
    for (int o = 8; o >= 1; o >>= 1) {
      dot += __shfl_xor(dot, o, 16);
      ns  += __shfl_xor(ns,  o, 16);
      nt  += __shfl_xor(nt,  o, 16);
    }
    if (k == 0) {
      float denom = sqrtf(ns) * sqrtf(nt);
      denom = denom > COS_EPS ? denom : COS_EPS;
      Alds[e * KPAD + 256] = (__bf16)(dot / denom);
    }
  }
  __syncthreads();
  // ee1: [16,288] x [288,128] -> h [16,128]; 8 waves, one 16x16 tile each
  v8f acc = {};
  for (int kc = 0; kc < 9; ++kc) {
    for (int i = t; i < 32 * 128; i += 256) {
      int kr = i >> 7, n = i & 127;
      int kg = kc * 32 + kr;
      Wlds[bswz(kr, n)] = (kg < 257) ? (__bf16)W1[kg * 128 + n] : (__bf16)0.f;
    }
    __syncthreads();
    v16bf a = load_a_frag(Alds, KPAD, kc * 32, lane);
    v16bf b = load_b_frag_sw(Wlds, wave, lane);
    acc = wmma_bf16(a, b, acc);
    __syncthreads();
  }
  {
    int n = lane & 15, kh = lane >> 4;
    int c = wave * 16 + n;
    float bias = B1[c];
#pragma unroll
    for (int r = 0; r < 8; ++r) Hlds[(r + 8 * kh) * 128 + c] = (__bf16)relu_f(acc[r] + bias);
  }
  __syncthreads();
  // ee2: [16,128] x [128,64]; 4 waves produce output, all help load W chunks
  v8f acc2 = {};
  for (int kc = 0; kc < 4; ++kc) {
    for (int i = t; i < 32 * 64; i += 256) {
      int kr = i >> 6, n = i & 63;
      Wlds[bswz(kr, n)] = (__bf16)W2[(kc * 32 + kr) * 64 + n];
    }
    __syncthreads();
    if (wave < 4) {
      v16bf a = load_a_frag(Hlds, 128, kc * 32, lane);
      v16bf b = load_b_frag_sw(Wlds, wave, lane);
      acc2 = wmma_bf16(a, b, acc2);
    }
    __syncthreads();
  }
  if (wave < 4) {
    int n = lane & 15, kh = lane >> 4;
    int c = wave * 16 + n;
    float bias = B2[c];
    float lsum = 0.f, lsq = 0.f;
#pragma unroll
    for (int r = 0; r < 8; ++r) {
      int m = r + 8 * kh;
      float v = acc2[r] + bias;
      epre[(long long)(e0 + m) * H + c] = v;
      lsum += v; lsq += v * v;
    }
    atomicAdd(&esum[c], lsum);
    atomicAdd(&esq[c], lsq);
  }
  __syncthreads();
  if (t < H) { atomicAdd(&stats[t], esum[t]); atomicAdd(&stats[H + t], esq[t]); }
}

// ---------------- edge BN apply + relu -> new edge_attr ----------------
__global__ __launch_bounds__(256)
void edge_bn_apply_kernel(const float* __restrict__ epre, const float* __restrict__ mi,
                          const float* __restrict__ g, const float* __restrict__ b,
                          float* __restrict__ eout) {
  const long long total = (long long)N_EDGES * H;
  for (long long i = (long long)blockIdx.x * 256 + threadIdx.x; i < total;
       i += (long long)gridDim.x * 256) {
    int c = (int)(i & 63);
    eout[i] = relu_f(g[c] * (epre[i] - mi[c]) * mi[H + c] + b[c]);
  }
}

extern "C" void kernel_launch(void* const* d_in, const int* in_sizes, int n_in,
                              void* d_out, int out_size, void* d_ws, size_t ws_size,
                              hipStream_t stream) {
  (void)in_sizes; (void)n_in; (void)out_size; (void)ws_size;
  const float* X  = (const float*)d_in[0];
  const int*   EI = (const int*)d_in[1];
  const float* EA = (const float*)d_in[2];
  // params pytree (sorted dict keys): blocks[0..1] then edge_proj{b,w} then node_proj{b,w}
  struct BlkP {
    const float *bn_edge_b, *bn_edge_g, *bn_node_b, *bn_node_g;
    const float *ee1_b, *ee1_w, *ee2_b, *ee2_w;
    const float *pdn_e1_b, *pdn_e1_w, *pdn_e2_b, *pdn_e2_w, *pdn_n_b, *pdn_n_w;
  } blk[2];
  for (int b = 0; b < 2; ++b) {
    int base = 3 + b * 14;
    blk[b].bn_edge_b = (const float*)d_in[base + 0];
    blk[b].bn_edge_g = (const float*)d_in[base + 1];
    blk[b].bn_node_b = (const float*)d_in[base + 2];
    blk[b].bn_node_g = (const float*)d_in[base + 3];
    blk[b].ee1_b     = (const float*)d_in[base + 4];
    blk[b].ee1_w     = (const float*)d_in[base + 5];
    blk[b].ee2_b     = (const float*)d_in[base + 6];
    blk[b].ee2_w     = (const float*)d_in[base + 7];
    blk[b].pdn_e1_b  = (const float*)d_in[base + 8];
    blk[b].pdn_e1_w  = (const float*)d_in[base + 9];
    blk[b].pdn_e2_b  = (const float*)d_in[base + 10];
    blk[b].pdn_e2_w  = (const float*)d_in[base + 11];
    blk[b].pdn_n_b   = (const float*)d_in[base + 12];
    blk[b].pdn_n_w   = (const float*)d_in[base + 13];
  }
  const float* edge_proj_b = (const float*)d_in[31];
  const float* edge_proj_w = (const float*)d_in[32];
  const float* node_proj_b = (const float*)d_in[33];
  const float* node_proj_w = (const float*)d_in[34];
  const int* rowp = EI;
  const int* colp = EI + N_EDGES;

  float* ws = (float*)d_ws;
  float* x_h     = ws; ws += (size_t)N_NODES * H;
  float* sbuf    = ws; ws += (size_t)N_NODES * H;
  float* agg     = ws; ws += (size_t)N_NODES * H;
  float* omega   = ws; ws += (size_t)N_EDGES;
  float* eattr_h = ws; ws += (size_t)N_EDGES * H;
  float* epre    = ws; ws += (size_t)N_EDGES * H;
  float* stats   = ws; ws += 640;
  // stats layout: [0..127] node sum/sq, [128..255] edge sum/sq,
  //               [256..383] node mean/inv, [384..511] edge mean/inv

  node_proj_kernel<<<N_NODES / 16, 128, 0, stream>>>(X, node_proj_w, node_proj_b, x_h);
  edge_proj_kernel<<<N_EDGES / 16, 128, 0, stream>>>(EA, edge_proj_w, edge_proj_b, eattr_h);

  for (int b = 0; b < 2; ++b) {
    zero_kernel<<<256, 256, 0, stream>>>(agg, (long long)N_NODES * H);
    zero_kernel<<<1, 256, 0, stream>>>(stats, 512);
    pdn_edge_kernel<<<N_EDGES / 16, 64, 0, stream>>>(eattr_h, blk[b].pdn_e1_w, blk[b].pdn_e1_b,
                                                     blk[b].pdn_e2_w, blk[b].pdn_e2_b, omega);
    pdn_node_scatter_kernel<<<N_EDGES / 16, 128, 0, stream>>>(x_h, rowp, colp, blk[b].pdn_n_w,
                                                              blk[b].pdn_n_b, omega, agg);
    node_sum_stats_kernel<<<512, 256, 0, stream>>>(x_h, agg, sbuf, stats);
    bn_finalize_kernel<<<1, 64, 0, stream>>>(stats, (float)N_NODES, stats + 256);
    node_bn_apply_kernel<<<512, 256, 0, stream>>>(sbuf, stats + 256, blk[b].bn_node_g,
                                                  blk[b].bn_node_b, x_h);
    edge_emb_kernel<<<N_EDGES / 16, 256, 0, stream>>>(eattr_h, x_h, rowp, colp, blk[b].ee1_w,
                                                      blk[b].ee1_b, blk[b].ee2_w, blk[b].ee2_b,
                                                      epre, stats + 128);
    bn_finalize_kernel<<<1, 64, 0, stream>>>(stats + 128, (float)N_EDGES, stats + 384);
    edge_bn_apply_kernel<<<1024, 256, 0, stream>>>(epre, stats + 384, blk[b].bn_edge_g,
                                                   blk[b].bn_edge_b, eattr_h);
  }
  (void)hipMemcpyAsync(d_out, x_h, (size_t)N_NODES * H * sizeof(float),
                       hipMemcpyDeviceToDevice, stream);
}
